// kernel_generated_0_62904091017456
// MI455X (gfx1250) — matrix operand order repaired and checked by reading, not compiled, not run on hardware
//
#include <hip/hip_runtime.h>

typedef __attribute__((ext_vector_type(16))) _Float16 v16h;
typedef __attribute__((ext_vector_type(8)))  _Float16 v8h;
typedef __attribute__((ext_vector_type(8)))  float    v8f;

#define TM        8                 // output m-rows per block
#define NSLOT     12                // x-row slots: 0..9 = m0-1..m0+8, 10/11 = wrap halo
#define XPITCH    72                // halves per (slot, n) row  (144B: 16B-aligned, conflict-friendly)
#define XSLOT     (64 * XPITCH)     // halves per slot
#define APITCH    200               // halves per A row (400B: 16B-aligned, conflict-friendly)
#define SMEM_HALVES (64 * APITCH + NSLOT * XSLOT)   // 12800 + 55296 = 68096

__global__ __launch_bounds__(512) void conv_shift_wmma(const float* __restrict__ x,
                                                       const float* __restrict__ W1,
                                                       const float* __restrict__ W2,
                                                       float* __restrict__ out)
{
    extern __shared__ _Float16 smem[];
    _Float16* ldsA = smem;                  // A[i][tap*64+j], pitch APITCH
    _Float16* ldsX = smem + 64 * APITCH;    // X[slot][n][j],  pitch XPITCH

    const int tid = threadIdx.x;
    const int bx  = blockIdx.x;
    const int l   = bx / (56 / TM);
    const int m0  = (bx % (56 / TM)) * TM;

    // ---- A = concat_k W2[:,k,:]  (f32 -> f16) ----
    for (int idx = tid; idx < 64 * 192; idx += 512) {
        const int i = idx / 192;
        const int r = idx - i * 192;
        ldsA[i * APITCH + r] = (_Float16)W2[i * 192 + r];
    }

    // ---- x tile -> LDS, (j,n)-transposed, f32 -> f16; OOB rows -> zeros ----
    for (int idx = tid; idx < NSLOT * 64 * 14; idx += 512) {
        const int u   = idx / (64 * 14);
        const int rem = idx - u * (64 * 14);
        const int j   = rem / 14;
        const int n4  = rem - j * 14;
        int g;
        if (u < 10)       g = m0 - 1 + u;                  // rows m0-1..m0+8
        else if (u == 10) g = (m0 == 0) ? 54 : m0 - 2;     // wrap-halo row(s)
        else              g = (m0 == 0) ? 55 : -1;         // slot 11 unused for m0>0
        float4 v = make_float4(0.f, 0.f, 0.f, 0.f);
        if (g >= 0 && g <= 55)
            v = *(const float4*)(x + ((size_t)l * 56 + g) * 3584 + j * 56 + n4 * 4);
        _Float16* dst = ldsX + u * XSLOT + (n4 * 4) * XPITCH + j;
        dst[0]          = (_Float16)v.x;
        dst[XPITCH]     = (_Float16)v.y;
        dst[2 * XPITCH] = (_Float16)v.z;
        dst[3 * XPITCH] = (_Float16)v.w;
    }
    __syncthreads();

    const int lane     = tid & 31;
    const int wave     = tid >> 5;    // 16 waves = 16 (i-tile, n-tile) combos
    const int laneN    = lane & 15;   // N column / M-row selector
    const int laneHalf = lane >> 4;   // 0: lanes 0-15, 1: lanes 16-31

    // s==0 tap -> slot (runtime-uniform selects, no branches)
    const int u0_t0 = 10;
    const int u0_t1 = (m0 == 0) ? 11 : 0;
    const int u0_t2 = (m0 == 0) ? 0  : 1;

    const int it = wave >> 2;
    const int nt = wave & 3;

    v8f acc[TM + 1];
    #pragma unroll
    for (int s = 0; s <= TM; ++s) acc[s] = (v8f){};

    const int arow  = it * 16 + laneN;
    const int kofsA = 8  * laneHalf;  // dense 16-bit A 16x32 lane layout
    const int kofsB = 8  * laneHalf;  // 16-bit B 32x16 lane layout
    const _Float16* bcol = ldsX + (nt * 16 + laneN) * XPITCH + kofsB;

    #pragma unroll
    for (int tap = 0; tap < 3; ++tap) {
        const int u0 = (tap == 0) ? u0_t0 : (tap == 1) ? u0_t1 : u0_t2;
        #pragma unroll
        for (int kb = 0; kb < 2; ++kb) {
            const _Float16* ap = ldsA + arow * APITCH + tap * 64 + kb * 32 + kofsA;
            const v8h alo = *(const v8h*)(ap);
            const v8h ahi = *(const v8h*)(ap + 16);
            const v16h a = __builtin_shufflevector(alo, ahi,
                                0,1,2,3,4,5,6,7,8,9,10,11,12,13,14,15);
            #pragma unroll
            for (int s = 0; s <= TM; ++s) {
                const int u = (s == 0) ? u0 : (s - 1 + tap);  // constant for s>=1
                const _Float16* bp = bcol + u * XSLOT + kb * 32;
                const v8h blo = *(const v8h*)(bp);
                const v8h bhi = *(const v8h*)(bp + 16);
                const v16h b = __builtin_shufflevector(blo, bhi,
                                    0,1,2,3,4,5,6,7,8,9,10,11,12,13,14,15);
                acc[s] = __builtin_amdgcn_wmma_f32_16x16x32_f16(
                             false, a, false, b, (short)0, acc[s], false, false);
            }
        }
    }

    // ---- epilogue: t6[m] = t4[m]*W1[i,0] + t4[m-1]*W1[i,1]; roll +1 along n ----
    const int n = nt * 16 + laneN;
    if (n < 56) {
        const int nOut = (n + 1) % 56;
        #pragma unroll
        for (int r = 0; r < 8; ++r) {
            const int i  = it * 16 + r + 8 * laneHalf;  // C-layout row
            const float w0 = W1[i * 2 + 0];
            const float w1 = W1[i * 2 + 1];
            #pragma unroll
            for (int s = 1; s <= TM; ++s) {
                const float t6 = acc[s][r] * w0 + acc[s - 1][r] * w1;
                const int m = m0 + s - 1;
                __builtin_nontemporal_store(
                    t6, out + (((size_t)l * 64 + i) * 56 + m) * 56 + nOut);
            }
        }
    }
}

extern "C" void kernel_launch(void* const* d_in, const int* in_sizes, int n_in,
                              void* d_out, int out_size, void* d_ws, size_t ws_size,
                              hipStream_t stream) {
    const float* x  = (const float*)d_in[0];
    const float* W1 = (const float*)d_in[1];
    const float* W2 = (const float*)d_in[2];
    float* out = (float*)d_out;

    const size_t shmem = (size_t)SMEM_HALVES * 2;   // bytes
    const dim3 grid(1024 * (56 / TM));              // 7168 blocks: (l, m-block)
    conv_shift_wmma<<<grid, 512, shmem, stream>>>(x, W1, W2, out);
}
